// SharedVAttention_24206435680297
// MI455X (gfx1250) — compile-verified, matrix operand order checked by reading, not run on hardware
//
#include <hip/hip_runtime.h>
#include <hip/hip_bf16.h>

// ---------------------------------------------------------------------------
// SharedVAttention for MI455X (gfx1250), wave32 + WMMA bf16 + TDM.
// Pipeline:
//   K0 wprep  : transpose+convert Wk,Wv (fp32) -> Wt bf16 [2048 cols][1024 d]
//   K1 qprep  : q = (q_in + bias) @ Wq * hd^-0.5   (RoPE at pos 0 == identity)
//   K2 kvgemm : fused K/V projection GEMM in bf16 WMMA, weight tiles streamed
//               by the Tensor Data Mover (tensor_load_to_lds + s_wait_tensorcnt),
//               fused RoPE(K) epilogue
//   K3 attn   : flash-decode, 8 N-splits per (b,h), online softmax partials
//   K4 reduce : combine split partials -> attn_out [B][1024] fp32
//   K5 oproj  : attn_out @ Wo + bo -> d_out
// ---------------------------------------------------------------------------

#define B_SZ   16
#define N_SZ   4096
#define D_SZ   1024
#define H_SZ   8
#define HD_SZ  128
#define NSPLIT 8
#define CHUNK  (N_SZ / NSPLIT)          // 512

typedef __attribute__((ext_vector_type(16))) __bf16 v16bf;
typedef __attribute__((ext_vector_type(8)))  float  v8f;
typedef unsigned int u32x4v __attribute__((ext_vector_type(4)));
typedef int          i32x8v __attribute__((ext_vector_type(8)));
typedef int          i32x4v __attribute__((ext_vector_type(4)));

union FragU { v16bf v; uint4 q[2]; };

__device__ __forceinline__ unsigned short f2bf(float f) {
  unsigned u = __builtin_bit_cast(unsigned, f);
  u += 0x7fffu + ((u >> 16) & 1u);            // round-to-nearest-even
  return (unsigned short)(u >> 16);
}
__device__ __forceinline__ unsigned pack2bf(float lo, float hi) {
  return (unsigned)f2bf(lo) | ((unsigned)f2bf(hi) << 16);
}
__device__ __forceinline__ float bflo(unsigned u) {
  return __builtin_bit_cast(float, u << 16);
}
__device__ __forceinline__ float bfhi(unsigned u) {
  return __builtin_bit_cast(float, u & 0xffff0000u);
}

// ---------------------------------------------------------------------------
// K0: Wt[c][d] = (c<1024) ? Wk[c/128][d][c%128] : Wv[d][c-1024]   (bf16)
// ---------------------------------------------------------------------------
__global__ void __launch_bounds__(256)
wprep_kernel(const float* __restrict__ Wk, const float* __restrict__ Wv,
             unsigned short* __restrict__ Wt) {
  __shared__ float tile[32][33];
  const int tx = threadIdx.x, ty = threadIdx.y;
  const int c0 = blockIdx.x * 32, d0 = blockIdx.y * 32;
#pragma unroll
  for (int i = 0; i < 4; ++i) {
    const int c = c0 + tx;
    const int d = d0 + ty + i * 8;
    float v;
    if (c < D_SZ)  v = Wk[(size_t)((c >> 7) * D_SZ + d) * HD_SZ + (c & 127)];
    else           v = Wv[(size_t)d * D_SZ + (c - D_SZ)];
    tile[ty + i * 8][tx] = v;
  }
  __syncthreads();
#pragma unroll
  for (int i = 0; i < 4; ++i) {
    const int c = c0 + ty + i * 8;
    const int d = d0 + tx;
    Wt[(size_t)c * D_SZ + d] = f2bf(tile[tx][ty + i * 8]);
  }
}

// ---------------------------------------------------------------------------
// K1: q_ws[b][h][kk] = scale * sum_d (q_in[b][d]+bias[b][d]) * Wq[h][d][kk]
// ---------------------------------------------------------------------------
__global__ void __launch_bounds__(128)
qprep_kernel(const float* __restrict__ q_in, const float* __restrict__ cbias,
             const float* __restrict__ Wq, float* __restrict__ q_ws) {
  __shared__ float qb[D_SZ];
  const int bh = blockIdx.x;             // b*H + h
  const int b = bh >> 3, h = bh & 7;
  const int kk = threadIdx.x;
  for (int i = threadIdx.x; i < D_SZ; i += 128)
    qb[i] = q_in[b * D_SZ + i] + cbias[b * D_SZ + i];
  __syncthreads();
  float acc = 0.f;
  const float* wq = Wq + (size_t)h * D_SZ * HD_SZ + kk;
  for (int d = 0; d < D_SZ; ++d)
    acc = fmaf(qb[d], wq[(size_t)d * HD_SZ], acc);
  q_ws[(size_t)bh * HD_SZ + kk] = acc * 0.08838834764831845f;   // 128^-0.5
}

// ---------------------------------------------------------------------------
// K2: fused K/V projection.  One WG = 128 token rows; A tile (128x1024 bf16)
// resident in LDS; sweep 16 column tiles; weight k-chunks streamed by TDM
// into a double-buffered, hardware-padded LDS stage.
// nt 0..7 -> K head nt (+RoPE); nt 8..15 -> V head nt-8.
// ---------------------------------------------------------------------------
#define A_STRIDE_B 2064                 // (1024+8) bf16 row stride, 4-bank skew
#define A_BYTES    (128 * A_STRIDE_B)   // 264192
#define W_STRIDE_B 80                   // 64B data + 16B TDM pad, 20-bank skew
#define W_BYTES    (128 * W_STRIDE_B)   // 10240
#define KV_SMEM    (A_BYTES + 2 * W_BYTES)

#if __has_builtin(__builtin_amdgcn_tensor_load_to_lds)
#define USE_TDM 1
#else
#define USE_TDM 0
#endif

extern __shared__ char smem[];

#if USE_TDM
// Per-wave TDM stage: 16-column slice of the weight tile, k-chunk kt.
// Tile in Wt (bf16, col-major rows of 1024): tile_dim0 = 32 elems (64B) along k,
// tile_dim1 = 16 cols, src stride = 1024 elems.  Hardware pads +16B after each
// 64B row stored -> 80B LDS column stride (matches fragment addressing).
__device__ __forceinline__ void tdm_stage_w(const unsigned short* Wt, int c0w,
                                            int kt, unsigned lds_off) {
  const unsigned long long ga =
      (unsigned long long)(uintptr_t)Wt +
      ((unsigned long long)c0w * (unsigned long long)D_SZ +
       (unsigned long long)kt * 32ull) * 2ull;
  u32x4v g0;
  g0[0] = 1u;                                  // count=1 user descriptor
  g0[1] = lds_off;                             // lds_addr (bytes)
  g0[2] = (unsigned)ga;                        // global_addr[31:0]
  g0[3] = (unsigned)(ga >> 32) | (2u << 30);   // global_addr[56:32] | type=2
  i32x8v g1;
  g1[0] = (1 << 16)      // data_size = 2 bytes
        | (1 << 20)      // pad_enable
        | (3 << 22)      // pad_interval: 16 DWORDs (64B) between pads
        | (3 << 25);     // pad_amount:   4 DWORDs (16B)
  g1[1] = (int)((unsigned)D_SZ << 16);         // tensor_dim0 = 1024 (lo16)
  g1[2] = (int)(2048u << 16);                  // tensor_dim1 = 2048 (lo16)
  g1[3] = (int)(32u << 16);                    // tile_dim0 = 32 elems
  g1[4] = 16;                                  // tile_dim1 = 16 cols, tile_dim2=0
  g1[5] = D_SZ;                                // tensor_dim0_stride = 1024 elems
  g1[6] = 0;
  g1[7] = 0;
  const i32x4v z4 = {0, 0, 0, 0};              // 2-D tensor: groups 2/3 unused
  const i32x8v z8 = {0, 0, 0, 0, 0, 0, 0, 0};
  __builtin_amdgcn_tensor_load_to_lds(g0, g1, z4, z4, z8, 0);
}
#endif

__global__ void __launch_bounds__(256)
kvgemm_kernel(const float* __restrict__ kv, const unsigned short* __restrict__ Wt,
              unsigned short* __restrict__ Kws, unsigned short* __restrict__ Vws) {
  char* ldsA = smem;
  char* ldsW = smem + A_BYTES;

  const int tid    = threadIdx.x;
  const int wave   = tid >> 5;
  const int lane   = tid & 31;
  const int laneLo = lane & 15;
  const int hi     = lane >> 4;                 // 0 / 1 half-wave
  const long r0    = (long)blockIdx.x * 128;    // global token row base
  const int  b     = (int)(r0 >> 12);           // / 4096
  const int  nbase = (int)(r0 & 4095);

  // LDS byte offset of the weight stage (flat LDS aperture keeps the byte
  // offset in addr[31:0], so the truncated generic pointer is the D# lds_addr)
  const unsigned ldsW_off = (unsigned)(uintptr_t)ldsW;

  // ---- stage A: 128 rows x 1024 fp32 -> bf16 LDS (coalesced float4 reads)
  const float4* kv4 = (const float4*)(kv + (size_t)r0 * D_SZ);
  for (int i = tid; i < 128 * 256; i += 256) {
    const int row = i >> 8, kq = i & 255;
    const float4 v = kv4[i];
    uint2 pk;
    pk.x = pack2bf(v.x, v.y);
    pk.y = pack2bf(v.z, v.w);
    *(uint2*)(ldsA + row * A_STRIDE_B + kq * 8) = pk;
  }
  __syncthreads();

  const int waveCol = wave * 16;                // per-wave TDM column slice
  const int colS = tid >> 1, halfS = tid & 1;   // manual-stage assignment

  for (int nt = 0; nt < 16; ++nt) {
    const int c0 = nt * 128;
    v8f acc[8] = {};

    // prologue: stage k-chunk 0 into buffer 0
#if USE_TDM
    tdm_stage_w(Wt, c0 + waveCol, 0, ldsW_off + waveCol * W_STRIDE_B);
    __builtin_amdgcn_s_wait_tensorcnt(0);
#else
    {
      const uint4* g = (const uint4*)(Wt + (size_t)(c0 + colS) * D_SZ + halfS * 16);
      const uint4 w0 = g[0], w1 = g[1];
      char* d = ldsW + colS * W_STRIDE_B + halfS * 32;
      *(uint4*)d = w0; *(uint4*)(d + 16) = w1;
    }
#endif
    __syncthreads();

    for (int kt = 0; kt < 32; ++kt) {
      const bool pref = (kt + 1) < 32;
#if USE_TDM
      if (pref)                                  // async DMA of next k-chunk
        tdm_stage_w(Wt, c0 + waveCol, kt + 1,
                    ldsW_off + ((kt + 1) & 1) * W_BYTES + waveCol * W_STRIDE_B);
#else
      uint4 w0, w1;
      if (pref) {
        const uint4* g = (const uint4*)(Wt + (size_t)(c0 + colS) * D_SZ +
                                        (kt + 1) * 32 + halfS * 16);
        w0 = g[0]; w1 = g[1];
      }
#endif
      char* wbuf = ldsW + (kt & 1) * W_BYTES;

      FragU a;                                    // A frag: 16x32 bf16
      {
        const char* ap = ldsA + (wave * 16 + laneLo) * A_STRIDE_B + kt * 64 + hi * 16;
        a.q[0] = *(const uint4*)ap;
        a.q[1] = *(const uint4*)(ap + 32);
      }
#pragma unroll
      for (int j = 0; j < 8; ++j) {               // B frags: 32x16 bf16 tiles
        FragU bf;
        const char* bp = wbuf + (16 * j + laneLo) * W_STRIDE_B + hi * 16;
        bf.q[0] = *(const uint4*)bp;
        bf.q[1] = *(const uint4*)(bp + 32);
        acc[j] = __builtin_amdgcn_wmma_f32_16x16x32_bf16(
            false, a.v, false, bf.v, (short)0, acc[j], false, false);
      }
#if USE_TDM
      __builtin_amdgcn_s_wait_tensorcnt(0);       // DMA overlapped the 8 WMMAs
#else
      if (pref) {
        char* d = ldsW + ((kt + 1) & 1) * W_BYTES + colS * W_STRIDE_B + halfS * 32;
        *(uint4*)d = w0; *(uint4*)(d + 16) = w1;
      }
#endif
      __syncthreads();
    }

    // ---- epilogue: C layout (vr,lane): row = 16*wave + vr + 8*hi, col = 16j + laneLo
    const int rowBase = wave * 16 + hi * 8;
    if (nt < 8) {                                 // K head nt, fused RoPE
      unsigned short* out = Kws + (size_t)(b * H_SZ + nt) * N_SZ * HD_SZ;
#pragma unroll
      for (int j = 0; j < 4; ++j) {
        const int kk1 = 16 * j + laneLo;          // < 64
        const float invf = __expf(-(float)kk1 * 0.10793367623409589f); // ln(1000)/64
#pragma unroll
        for (int vr = 0; vr < 8; ++vr) {
          const int n = nbase + rowBase + vr;     // token position
          const float ang = (float)n * invf;
          const float cs = __cosf(ang), sn = __sinf(ang);
          const float x1 = acc[j][vr], x2 = acc[j + 4][vr];
          out[(size_t)n * HD_SZ + kk1]      = f2bf(x1 * cs - x2 * sn);
          out[(size_t)n * HD_SZ + kk1 + 64] = f2bf(x2 * cs + x1 * sn);
        }
      }
    } else {                                      // V head nt-8
      unsigned short* out = Vws + (size_t)(b * H_SZ + (nt - 8)) * N_SZ * HD_SZ;
#pragma unroll
      for (int j = 0; j < 8; ++j) {
        const int kk = 16 * j + laneLo;
#pragma unroll
        for (int vr = 0; vr < 8; ++vr) {
          const int n = nbase + rowBase + vr;
          out[(size_t)n * HD_SZ + kk] = f2bf(acc[j][vr]);
        }
      }
    }
  }
}

// ---------------------------------------------------------------------------
// K3: flash-decode split.  grid=(B*H, NSPLIT), block=256.
// ---------------------------------------------------------------------------
__global__ void __launch_bounds__(256)
attn_kernel(const float* __restrict__ q_ws,
            const unsigned short* __restrict__ Kws,
            const unsigned short* __restrict__ Vws,
            float* __restrict__ part_o, float* __restrict__ part_ml) {
  __shared__ float qs[HD_SZ];
  __shared__ float sc[CHUNK];
  __shared__ float wred[8];
  __shared__ float tmp[256];

  const int bh = blockIdx.x, split = blockIdx.y;
  const int n0 = split * CHUNK;
  const int tid = threadIdx.x;
  const int wave = tid >> 5, lane = tid & 31;
  const size_t base = (size_t)bh * N_SZ * HD_SZ;
  const unsigned short* Kp = Kws + base;
  const unsigned short* Vp = Vws + base;

  if (tid < HD_SZ) qs[tid] = q_ws[(size_t)bh * HD_SZ + tid];   // pre-scaled q
  __syncthreads();

  // scores: one token per thread, 2 passes
#pragma unroll
  for (int it = 0; it < 2; ++it) {
    const int n = it * 256 + tid;
    const uint4* kp = (const uint4*)(Kp + (size_t)(n0 + n) * HD_SZ);
    float acc = 0.f;
#pragma unroll
    for (int i = 0; i < 16; ++i) {
      const uint4 u = kp[i];
      acc = fmaf(qs[8 * i + 0], bflo(u.x), acc);
      acc = fmaf(qs[8 * i + 1], bfhi(u.x), acc);
      acc = fmaf(qs[8 * i + 2], bflo(u.y), acc);
      acc = fmaf(qs[8 * i + 3], bfhi(u.y), acc);
      acc = fmaf(qs[8 * i + 4], bflo(u.z), acc);
      acc = fmaf(qs[8 * i + 5], bfhi(u.z), acc);
      acc = fmaf(qs[8 * i + 6], bflo(u.w), acc);
      acc = fmaf(qs[8 * i + 7], bfhi(u.w), acc);
    }
    sc[n] = acc;
  }
  __syncthreads();

  // chunk max
  float m = fmaxf(sc[tid], sc[tid + 256]);
#pragma unroll
  for (int off = 16; off >= 1; off >>= 1) m = fmaxf(m, __shfl_xor(m, off, 32));
  if (lane == 0) wred[wave] = m;
  __syncthreads();
  m = wred[0];
#pragma unroll
  for (int w = 1; w < 8; ++w) m = fmaxf(m, wred[w]);
  __syncthreads();

  // exp + sum
  const float p0 = __expf(sc[tid] - m);
  const float p1 = __expf(sc[tid + 256] - m);
  __syncthreads();
  sc[tid] = p0; sc[tid + 256] = p1;
  float l = p0 + p1;
#pragma unroll
  for (int off = 16; off >= 1; off >>= 1) l += __shfl_xor(l, off, 32);
  if (lane == 0) wred[wave] = l;
  __syncthreads();
  l = 0.f;
#pragma unroll
  for (int w = 0; w < 8; ++w) l += wred[w];

  // PV: thread (k = tid&127, half = tid>>7) accumulates 256 tokens
  const int k = tid & 127, half = tid >> 7;
  float acc = 0.f;
  const int nA = half * 256;
  for (int n = nA; n < nA + 256; ++n)
    acc = fmaf(sc[n], bflo((unsigned)Vp[(size_t)(n0 + n) * HD_SZ + k]), acc);
  tmp[tid] = acc;
  __syncthreads();
  if (tid < 128) {
    part_o[(size_t)(bh * NSPLIT + split) * HD_SZ + tid] = tmp[tid] + tmp[tid + 128];
  }
  if (tid == 0) {
    part_ml[(bh * NSPLIT + split) * 2 + 0] = m;
    part_ml[(bh * NSPLIT + split) * 2 + 1] = l;
  }
}

// ---------------------------------------------------------------------------
// K4: combine splits -> attn_out[b][h*128+k]
// ---------------------------------------------------------------------------
__global__ void __launch_bounds__(128)
reduce_kernel(const float* __restrict__ part_o, const float* __restrict__ part_ml,
              float* __restrict__ attn_out) {
  const int bh = blockIdx.x;
  const int k = threadIdx.x;
  float M = -3.4e38f;
#pragma unroll
  for (int s = 0; s < NSPLIT; ++s)
    M = fmaxf(M, part_ml[(bh * NSPLIT + s) * 2]);
  float denom = 0.f, acc = 0.f;
#pragma unroll
  for (int s = 0; s < NSPLIT; ++s) {
    const float w = __expf(part_ml[(bh * NSPLIT + s) * 2] - M);
    denom = fmaf(part_ml[(bh * NSPLIT + s) * 2 + 1], w, denom);
    acc = fmaf(w, part_o[(size_t)(bh * NSPLIT + s) * HD_SZ + k], acc);
  }
  attn_out[(size_t)bh * HD_SZ + k] = acc / denom;   // [b][h*128+k] contiguous
}

// ---------------------------------------------------------------------------
// K5: out = attn_out @ Wo + bo.  grid=(4,B), block=256.
// ---------------------------------------------------------------------------
__global__ void __launch_bounds__(256)
oproj_kernel(const float* __restrict__ attn_out, const float* __restrict__ Wo,
             const float* __restrict__ bo, float* __restrict__ out) {
  __shared__ float a[D_SZ];
  const int b = blockIdx.y;
  for (int i = threadIdx.x; i < D_SZ; i += 256)
    a[i] = attn_out[b * D_SZ + i];
  __syncthreads();
  const int e = blockIdx.x * 256 + threadIdx.x;
  float acc = 0.f;
  for (int d = 0; d < D_SZ; ++d)
    acc = fmaf(a[d], Wo[(size_t)d * D_SZ + e], acc);
  out[b * D_SZ + e] = acc + bo[e];
}

// ---------------------------------------------------------------------------
extern "C" void kernel_launch(void* const* d_in, const int* in_sizes, int n_in,
                              void* d_out, int out_size, void* d_ws, size_t ws_size,
                              hipStream_t stream) {
  const float* q_in  = (const float*)d_in[0];
  const float* kv_in = (const float*)d_in[1];
  const float* cbias = (const float*)d_in[2];
  const float* Wq    = (const float*)d_in[3];
  const float* Wk    = (const float*)d_in[4];
  const float* Wv    = (const float*)d_in[5];
  const float* Wo    = (const float*)d_in[6];
  const float* bo    = (const float*)d_in[7];
  float* out = (float*)d_out;

  // workspace carve-up
  char* ws = (char*)d_ws;
  unsigned short* Wt      = (unsigned short*)ws;                   // 4 MB
  float*          q_ws    = (float*)(ws + (size_t)4 * 1024 * 1024);
  unsigned short* Kws     = (unsigned short*)(ws + (size_t)4 * 1024 * 1024 + 65536);
  unsigned short* Vws     = (unsigned short*)((char*)Kws + (size_t)B_SZ * H_SZ * N_SZ * HD_SZ * 2);
  float*          part_o  = (float*)((char*)Vws + (size_t)B_SZ * H_SZ * N_SZ * HD_SZ * 2);
  float*          part_ml = part_o + (size_t)B_SZ * H_SZ * NSPLIT * HD_SZ;
  float*          attn_o  = part_ml + (size_t)B_SZ * H_SZ * NSPLIT * 2;

  (void)hipFuncSetAttribute((const void*)kvgemm_kernel,
                            hipFuncAttributeMaxDynamicSharedMemorySize, KV_SMEM);

  wprep_kernel<<<dim3(2048 / 32, 1024 / 32), dim3(32, 8), 0, stream>>>(Wk, Wv, Wt);
  qprep_kernel<<<dim3(B_SZ * H_SZ), dim3(128), 0, stream>>>(q_in, cbias, Wq, q_ws);
  kvgemm_kernel<<<dim3((B_SZ * N_SZ) / 128), dim3(256), KV_SMEM, stream>>>(kv_in, Wt, Kws, Vws);
  attn_kernel<<<dim3(B_SZ * H_SZ, NSPLIT), dim3(256), 0, stream>>>(q_ws, Kws, Vws, part_o, part_ml);
  reduce_kernel<<<dim3(B_SZ * H_SZ), dim3(128), 0, stream>>>(part_o, part_ml, attn_o);
  oproj_kernel<<<dim3(4, B_SZ), dim3(256), 0, stream>>>(attn_o, Wo, bo, out);
}
